// MultiHeadAttention_88227218194919
// MI455X (gfx1250) — matrix operand order repaired and checked by reading, not compiled, not run on hardware
//
#include <hip/hip_runtime.h>
#include <hip/hip_bf16.h>

// MI455X (gfx1250) multi-head attention:
//   f32 -> f16 convert, WMMA f16 GEMM projections (async-LDS double-buffered),
//   flash attention with base-2 online softmax (WMMA for QK^T and PV, K/V tiles
//   staged via GLOBAL_LOAD_ASYNC_TO_LDS_B128), WMMA f16 output projection.
// Workspace requirement: ~120 MB + 16 KB (see layout in kernel_launch).

typedef __attribute__((ext_vector_type(16))) _Float16     v16h;
typedef __attribute__((ext_vector_type(8)))  float        v8f;
typedef __attribute__((ext_vector_type(4)))  float        v4f;
typedef __attribute__((ext_vector_type(4)))  _Float16     v4h;
typedef __attribute__((ext_vector_type(4)))  unsigned int v4u;

#define D_MODEL 1024
#define NHEAD   16
#define DKH     64
#define SEQ     2048
#define BATCH   4

// 1/sqrt(64) * log2(e): projection scale so softmax runs natively in base 2
#define QSCALE  (0.125f * 1.44269504088896340736f)

union FragU { v16h h; v4u u[2]; };

// Low 32 bits of a generic pointer to __shared__ == LDS byte address
// (flat->LDS mapping uses addr[31:0], ISA 10.2).
static __device__ __forceinline__ unsigned lds_off(const void* p) {
  return (unsigned)(uintptr_t)p;
}

// GLOBAL_LOAD_ASYNC_TO_LDS_B128: memory -> LDS, tracked with ASYNCcnt.
static __device__ __forceinline__ void async_ld_b128(unsigned lds, const void* g) {
  asm volatile("global_load_async_to_lds_b128 %0, %1, off"
               :: "v"(lds), "v"((unsigned long long)(uintptr_t)g)
               : "memory");
}
static __device__ __forceinline__ void wait_async0() {
  asm volatile("s_wait_asynccnt 0x0" ::: "memory");
}

// A-matrix fragment (16x32 f16, M x K). ISA 7.12.2:
// lane<16: row=lane, k in {0..7, 16..23}; lane>=16: row=lane-16, k in {8..15, 24..31}
static __device__ __forceinline__ v16h load_frag_a(const _Float16* base, int rowStride) {
  const int lane = threadIdx.x & 31;
  const int r    = lane & 15;
  const int kb   = (lane >> 4) << 3;
  const _Float16* p = base + r * rowStride + kb;
  FragU f;
  f.u[0] = *reinterpret_cast<const v4u*>(p);
  f.u[1] = *reinterpret_cast<const v4u*>(p + 16);
  return f.h;
}

// B-matrix fragment (32x16 f16, K x N), source stored N-major (16 rows x 32 k).
// lane<16: col=lane, k in {0..7, 16..23}; lane>=16: col=lane-16, k in {8..15, 24..31}
static __device__ __forceinline__ v16h load_frag_b(const _Float16* base, int rowStride) {
  const int lane = threadIdx.x & 31;
  const int r    = lane & 15;
  const int kb   = (lane >> 4) << 3;
  const _Float16* p = base + r * rowStride + kb;
  FragU f;
  f.u[0] = *reinterpret_cast<const v4u*>(p);
  f.u[1] = *reinterpret_cast<const v4u*>(p + 16);
  return f.h;
}

static __device__ __forceinline__ float hmax16(float v) {
  v = fmaxf(v, __shfl_xor(v, 1, 32));
  v = fmaxf(v, __shfl_xor(v, 2, 32));
  v = fmaxf(v, __shfl_xor(v, 4, 32));
  v = fmaxf(v, __shfl_xor(v, 8, 32));
  return v;
}
static __device__ __forceinline__ float hsum16(float v) {
  v += __shfl_xor(v, 1, 32);
  v += __shfl_xor(v, 2, 32);
  v += __shfl_xor(v, 4, 32);
  v += __shfl_xor(v, 8, 32);
  return v;
}

#define WMMA_F16(A, B, C) \
  __builtin_amdgcn_wmma_f32_16x16x32_f16(false, (A), false, (B), (short)0, (C), false, false)

// ---------------- f32 -> f16 conversion ----------------
__global__ __launch_bounds__(256) void cvt_f16(const float* __restrict__ in,
                                               _Float16* __restrict__ out, int n4) {
  int i = blockIdx.x * 256 + threadIdx.x;
  if (i >= n4) return;
  v4f x = reinterpret_cast<const v4f*>(in)[i];
  v4h y;
  y[0] = (_Float16)x[0]; y[1] = (_Float16)x[1];
  y[2] = (_Float16)x[2]; y[3] = (_Float16)x[3];
  reinterpret_cast<v4h*>(out)[i] = y;
}

// f32 [K][N] -> f16 [N][K] (weight transpose so GEMM B-fragments are contiguous)
__global__ __launch_bounds__(256) void cvt_f16_T(const float* __restrict__ in,
                                                 _Float16* __restrict__ out) {
  int i = blockIdx.x * 256 + threadIdx.x;   // over D_MODEL*D_MODEL
  int k = i >> 10;
  int n = i & (D_MODEL - 1);
  out[n * D_MODEL + k] = (_Float16)in[i];
}

// ---------------- mask tile reduction ----------------
// flags[b][qt][kt] = 1 iff entire 64x64 mask tile is non-zero.
__global__ __launch_bounds__(256) void mask_tile_reduce(const int* __restrict__ mask,
                                                        int* __restrict__ flags) {
  const int b = blockIdx.z, qt = blockIdx.y, kt = blockIdx.x;
  __shared__ int ok;
  if (threadIdx.x == 0) ok = 1;
  __syncthreads();
  const int row = qt * 64 + (threadIdx.x >> 2);       // 64 rows, 4 threads/row
  const int col = kt * 64 + (threadIdx.x & 3) * 16;   // 16 ints each
  const int* p = mask + ((size_t)b * SEQ + row) * SEQ + col;
  int all = 1;
#pragma unroll
  for (int c = 0; c < 4; ++c) {
    v4u m = reinterpret_cast<const v4u*>(p)[c];
    all &= (m[0] != 0u) & (m[1] != 0u) & (m[2] != 0u) & (m[3] != 0u);
  }
  if (!all) ok = 0;
  __syncthreads();
  if (threadIdx.x == 0) flags[((size_t)b * 32 + qt) * 32 + kt] = ok;
}

// ---------------- WMMA GEMM: C = A(f16 MxK) * Bt(f16 NxK)^T ----------------
// Double-buffered LDS staging through GLOBAL_LOAD_ASYNC_TO_LDS_B128.
// MODE 0: f16 out scattered to [B,H,S,DK]   (q, k projections)
// MODE 1: f16 out scattered to [B,H,DK,S]   (v projection, transposed)
// MODE 2: f32 out row-major [M,N]           (final output)
template <int MODE>
__global__ __launch_bounds__(256) void gemm_f16(const _Float16* __restrict__ A,
                                                const _Float16* __restrict__ Bt,
                                                void* __restrict__ Cout,
                                                int M, int N, int K, float scale) {
  __shared__ _Float16 lA[2][128 * 32];
  __shared__ _Float16 lB[2][64 * 32];
  const int tid  = threadIdx.x;
  const int wave = tid >> 5;
  const int lane = tid & 31;
  const int mBase = blockIdx.y * 128;
  const int nBase = blockIdx.x * 64;

  // async-stage one 32-wide k-slice (A: 128x32, Bt: 64x32) into buffer `buf`
  auto stage = [&](int buf, int k0) {
    const int r = tid >> 1;             // 0..127
    const int c = (tid & 1) * 16;       // 0 or 16 (f16 elems)
    const _Float16* srcA = A + (size_t)(mBase + r) * K + k0 + c;
    const unsigned  la   = lds_off(&lA[buf][r * 32 + c]);
    async_ld_b128(la,      srcA);
    async_ld_b128(la + 16, srcA + 8);
    const int rb = tid >> 2;            // 0..63
    const int cb = (tid & 3) * 8;       // 16B chunks
    async_ld_b128(lds_off(&lB[buf][rb * 32 + cb]),
                  Bt + (size_t)(nBase + rb) * K + k0 + cb);
  };

  v8f acc[4] = {};

  stage(0, 0);
  for (int k0 = 0; k0 < K; k0 += 32) {
    const int cur = (k0 >> 5) & 1;
    wait_async0();
    __syncthreads();
    if (k0 + 32 < K) stage(cur ^ 1, k0 + 32);   // overlap next fetch with compute

    // load all fragments first so DS latency overlaps WMMA issue
    v16h a  = load_frag_a(&lA[cur][(wave * 16) * 32], 32);
    v16h b0 = load_frag_b(&lB[cur][ 0 * 32], 32);
    v16h b1 = load_frag_b(&lB[cur][16 * 32], 32);
    v16h b2 = load_frag_b(&lB[cur][32 * 32], 32);
    v16h b3 = load_frag_b(&lB[cur][48 * 32], 32);
    acc[0] = WMMA_F16(a, b0, acc[0]);
    acc[1] = WMMA_F16(a, b1, acc[1]);
    acc[2] = WMMA_F16(a, b2, acc[2]);
    acc[3] = WMMA_F16(a, b3, acc[3]);
  }

  // epilogue (C layout: VGPR j -> row j + 8*(lane>=16); col = lane&15)
  const int row0 = (lane >> 4) * 8;
  const int col  = lane & 15;
#pragma unroll
  for (int nt = 0; nt < 4; ++nt) {
#pragma unroll
    for (int j = 0; j < 8; ++j) {
      const int m = mBase + wave * 16 + j + row0;
      const int n = nBase + nt * 16 + col;
      const float val = acc[nt][j] * scale;
      if (MODE == 2) {
        reinterpret_cast<float*>(Cout)[(size_t)m * N + n] = val;
      } else {
        const int b = m >> 11, s = m & (SEQ - 1);
        const int h = n >> 6,  d = n & (DKH - 1);
        _Float16* o = reinterpret_cast<_Float16*>(Cout);
        if (MODE == 0)
          o[(((size_t)(b * NHEAD + h) * SEQ) + s) * DKH + d] = (_Float16)val;
        else
          o[(((size_t)(b * NHEAD + h) * DKH) + d) * SEQ + s] = (_Float16)val;
      }
    }
  }
}

// ---------------- flash attention ----------------
// grid: (B*H, SEQ/64), block: 128 threads (4 waves, 16 queries each)
// q,k: [B,H,S,DK] f16 (q pre-scaled by log2(e)/sqrt(DK)); vT: [B,H,DK,S] f16
// K/V 64x64 tiles are async-staged into double-buffered LDS (shared by all
// 4 waves -> 4x less L2 traffic than per-wave global fragment loads).
// Softmax runs in base 2 (v_exp_f32 is natively 2^x).
__global__ __launch_bounds__(128) void flash_attn(const _Float16* __restrict__ q,
                                                  const _Float16* __restrict__ k,
                                                  const _Float16* __restrict__ vT,
                                                  const int* __restrict__ mask,
                                                  const int* __restrict__ flags,
                                                  _Float16* __restrict__ out) {
  __shared__ _Float16 lK[2][64 * 64];     // [key][d]
  __shared__ _Float16 lV[2][64 * 64];     // [d][key]
  __shared__ _Float16 ldsP[4][16 * 64];   // per-wave P tile (C->A layout bounce)

  const int lane = threadIdx.x & 31;
  const int wave = threadIdx.x >> 5;
  const int bh   = blockIdx.x;
  const int b    = bh >> 4;
  const int h    = bh & (NHEAD - 1);
  const int qBase = blockIdx.y * 64 + wave * 16;

  const _Float16* qp = q  + (size_t)bh * SEQ * DKH;
  const _Float16* kp = k  + (size_t)bh * SEQ * DKH;
  const _Float16* vp = vT + (size_t)bh * DKH * SEQ;
  const int* mp      = mask  + (size_t)b * SEQ * SEQ;
  const int* fp      = flags + ((size_t)b * 32 + blockIdx.y) * 32;

  // async-stage 64x64 K and V tiles for key block `kb` into buffer `buf`
  auto stageKV = [&](int buf, int kb) {
    const int r = threadIdx.x >> 1;        // 0..63
    const int c = (threadIdx.x & 1) * 32;  // 64B halves (f16 elems)
    const _Float16* gk = kp + (size_t)(kb + r) * DKH + c;
    const _Float16* gv = vp + (size_t)r * SEQ + kb + c;
    const unsigned lk = lds_off(&lK[buf][r * 64 + c]);
    const unsigned lv = lds_off(&lV[buf][r * 64 + c]);
#pragma unroll
    for (int q8 = 0; q8 < 4; ++q8) {       // 4 x 16B each
      async_ld_b128(lk + q8 * 16, gk + q8 * 8);
      async_ld_b128(lv + q8 * 16, gv + q8 * 8);
    }
  };

  // Q fragments (held for entire kernel)
  v16h qa[2];
#pragma unroll
  for (int kk = 0; kk < 2; ++kk)
    qa[kk] = load_frag_a(qp + (size_t)qBase * DKH + kk * 32, DKH);

  const int row0 = (lane >> 4) * 8;
  const int col  = lane & 15;

  float m_[8], l_[8];
#pragma unroll
  for (int j = 0; j < 8; ++j) { m_[j] = -1e30f; l_[j] = 0.0f; }
  v8f o[4] = {};

  stageKV(0, 0);
  for (int kb = 0; kb < SEQ; kb += 64) {
    const int cur = (kb >> 6) & 1;
    wait_async0();
    __syncthreads();
    if (kb + 64 < SEQ) stageKV(cur ^ 1, kb + 64);   // overlap next K/V fetch

    // ---- scores S = (log2e/sqrt(dk)) * Q @ K^T  (16 x 64) ----
    v8f sc[4] = {};
#pragma unroll
    for (int kk = 0; kk < 2; ++kk) {
      v16h b0 = load_frag_b(&lK[cur][ 0 * 64 + kk * 32], 64);
      v16h b1 = load_frag_b(&lK[cur][16 * 64 + kk * 32], 64);
      v16h b2 = load_frag_b(&lK[cur][32 * 64 + kk * 32], 64);
      v16h b3 = load_frag_b(&lK[cur][48 * 64 + kk * 32], 64);
      sc[0] = WMMA_F16(qa[kk], b0, sc[0]);
      sc[1] = WMMA_F16(qa[kk], b1, sc[1]);
      sc[2] = WMMA_F16(qa[kk], b2, sc[2]);
      sc[3] = WMMA_F16(qa[kk], b3, sc[3]);
    }

    // ---- mask (fast path: tile known all-ones; flag is wave-uniform) ----
    const int flagv = __builtin_amdgcn_readfirstlane(fp[kb >> 6]);
    if (flagv == 0) {
#pragma unroll
      for (int nt = 0; nt < 4; ++nt) {
        const int key = kb + nt * 16 + col;
#pragma unroll
        for (int j = 0; j < 8; ++j) {
          const int qr = qBase + j + row0;
          const float cur_s = sc[nt][j];
          sc[nt][j] = (mp[(size_t)qr * SEQ + key] == 0) ? -1e9f : cur_s;
        }
      }
    }

    // ---- online softmax update (base 2) ----
    float alpha[8];
#pragma unroll
    for (int j = 0; j < 8; ++j) {
      float v = fmaxf(fmaxf(sc[0][j], sc[1][j]), fmaxf(sc[2][j], sc[3][j]));
      v = hmax16(v);
      const float mn = fmaxf(m_[j], v);
      alpha[j] = __builtin_amdgcn_exp2f(m_[j] - mn);
      m_[j] = mn;
    }
    float rs[8] = {0.f, 0.f, 0.f, 0.f, 0.f, 0.f, 0.f, 0.f};
#pragma unroll
    for (int nt = 0; nt < 4; ++nt) {
#pragma unroll
      for (int j = 0; j < 8; ++j) {
        const float p = __builtin_amdgcn_exp2f(sc[nt][j] - m_[j]);
        rs[j] += p;
        ldsP[wave][(j + row0) * 64 + nt * 16 + col] = (_Float16)p;
      }
    }
#pragma unroll
    for (int j = 0; j < 8; ++j) {
      l_[j] = l_[j] * alpha[j] + hsum16(rs[j]);
#pragma unroll
      for (int nt = 0; nt < 4; ++nt) {
        float oo = o[nt][j];
        o[nt][j] = oo * alpha[j];
      }
    }

    // ---- O += P @ V ----
    __builtin_amdgcn_wave_barrier();   // order DS writes before DS reads (same wave)
#pragma unroll
    for (int kk = 0; kk < 2; ++kk) {
      v16h pa = load_frag_a(&ldsP[wave][kk * 32], 64);
      v16h b0 = load_frag_b(&lV[cur][ 0 * 64 + kk * 32], 64);
      v16h b1 = load_frag_b(&lV[cur][16 * 64 + kk * 32], 64);
      v16h b2 = load_frag_b(&lV[cur][32 * 64 + kk * 32], 64);
      v16h b3 = load_frag_b(&lV[cur][48 * 64 + kk * 32], 64);
      o[0] = WMMA_F16(pa, b0, o[0]);
      o[1] = WMMA_F16(pa, b1, o[1]);
      o[2] = WMMA_F16(pa, b2, o[2]);
      o[3] = WMMA_F16(pa, b3, o[3]);
    }
    __builtin_amdgcn_wave_barrier();   // WAR: next iteration overwrites ldsP
  }

  // ---- normalize and write out [B,S,D_MODEL] f16 ----
#pragma unroll
  for (int j = 0; j < 8; ++j) {
    const float inv = 1.0f / l_[j];
    const int s = qBase + j + row0;
#pragma unroll
    for (int nt = 0; nt < 4; ++nt) {
      out[((size_t)(b * SEQ + s)) * D_MODEL + h * DKH + nt * 16 + col] =
          (_Float16)(o[nt][j] * inv);
    }
  }
}

// ---------------- host-side launcher ----------------
extern "C" void kernel_launch(void* const* d_in, const int* in_sizes, int n_in,
                              void* d_out, int out_size, void* d_ws, size_t ws_size,
                              hipStream_t stream) {
  const float* Q   = (const float*)d_in[0];
  const float* K   = (const float*)d_in[1];
  const float* V   = (const float*)d_in[2];
  const int*   msk = (const int*)  d_in[3];
  const float* W_Q = (const float*)d_in[4];
  const float* W_K = (const float*)d_in[5];
  const float* W_V = (const float*)d_in[6];
  const float* W_O = (const float*)d_in[7];

  const size_t NTOK = (size_t)BATCH * SEQ;          // 8192 rows
  const size_t NQ   = NTOK * D_MODEL;               // 8.39M elems
  const size_t ND   = (size_t)D_MODEL * D_MODEL;    // 1.05M elems

  char* ws = (char*)d_ws;
  size_t off = 0;
  _Float16* hQ    = (_Float16*)(ws + off); off += NQ * 2;        // 16 MB
  _Float16* hK    = (_Float16*)(ws + off); off += NQ * 2;
  _Float16* hV    = (_Float16*)(ws + off); off += NQ * 2;
  _Float16* hWqT  = (_Float16*)(ws + off); off += ND * 2;        // 2 MB each
  _Float16* hWkT  = (_Float16*)(ws + off); off += ND * 2;
  _Float16* hWvT  = (_Float16*)(ws + off); off += ND * 2;
  _Float16* hWoT  = (_Float16*)(ws + off); off += ND * 2;
  _Float16* qbuf  = (_Float16*)(ws + off); off += NQ * 2;        // [B,H,S,DK]
  _Float16* kbuf  = (_Float16*)(ws + off); off += NQ * 2;        // [B,H,S,DK]
  _Float16* vbufT = (_Float16*)(ws + off); off += NQ * 2;        // [B,H,DK,S]
  _Float16* hO    = (_Float16*)(ws + off); off += NQ * 2;        // [B,S,D]
  int*      flags = (int*)(ws + off);      off += (size_t)BATCH * 32 * 32 * 4;

  // 1) convert inputs / weights (weights transposed)
  const int n4 = (int)(NQ / 4);
  cvt_f16<<<(n4 + 255) / 256, 256, 0, stream>>>(Q, hQ, n4);
  cvt_f16<<<(n4 + 255) / 256, 256, 0, stream>>>(K, hK, n4);
  cvt_f16<<<(n4 + 255) / 256, 256, 0, stream>>>(V, hV, n4);
  cvt_f16_T<<<(int)(ND / 256), 256, 0, stream>>>(W_Q, hWqT);
  cvt_f16_T<<<(int)(ND / 256), 256, 0, stream>>>(W_K, hWkT);
  cvt_f16_T<<<(int)(ND / 256), 256, 0, stream>>>(W_V, hWvT);
  cvt_f16_T<<<(int)(ND / 256), 256, 0, stream>>>(W_O, hWoT);

  // 2) mask tile flags
  mask_tile_reduce<<<dim3(32, 32, BATCH), 256, 0, stream>>>(msk, flags);

  // 3) projections (q gets log2e/sqrt(dk) folded in for base-2 softmax)
  dim3 ggrid(D_MODEL / 64, (unsigned)(NTOK / 128));
  gemm_f16<0><<<ggrid, 256, 0, stream>>>(hQ, hWqT, qbuf, (int)NTOK, D_MODEL, D_MODEL, QSCALE);
  gemm_f16<0><<<ggrid, 256, 0, stream>>>(hK, hWkT, kbuf, (int)NTOK, D_MODEL, D_MODEL, 1.0f);
  gemm_f16<1><<<ggrid, 256, 0, stream>>>(hV, hWvT, vbufT, (int)NTOK, D_MODEL, D_MODEL, 1.0f);

  // 4) flash attention
  flash_attn<<<dim3(BATCH * NHEAD, SEQ / 64), 128, 0, stream>>>(
      qbuf, kbuf, vbufT, msk, flags, hO);

  // 5) output projection -> f32 d_out
  gemm_f16<2><<<ggrid, 256, 0, stream>>>(hO, hWoT, d_out, (int)NTOK, D_MODEL, D_MODEL, 1.0f);
}
